// Head_27152783245814
// MI455X (gfx1250) — matrix operand order repaired and checked by reading, not compiled, not run on hardware
//
#include <hip/hip_runtime.h>

#define EMBD 1024
#define HEAD 128
#define SEQT 2048
#define NB   8

typedef __attribute__((ext_vector_type(16))) __bf16          v16bf;
typedef __attribute__((ext_vector_type(16))) unsigned short  v16u;
typedef __attribute__((ext_vector_type(8)))  unsigned short  v8u;
typedef __attribute__((ext_vector_type(8)))  float           v8f;

__device__ __forceinline__ unsigned short f2bf(float f) {
  union { float f; unsigned int u; } v; v.f = f;
  unsigned int u = v.u;
  u += 0x7fffu + ((u >> 16) & 1u);   // round-to-nearest-even
  return (unsigned short)(u >> 16);
}
__device__ __forceinline__ unsigned int pk2bf(float a, float b) {
  return (unsigned int)f2bf(a) | ((unsigned int)f2bf(b) << 16);
}
__device__ __forceinline__ v16u cat16(v8u lo, v8u hi) {
  return __builtin_shufflevector(lo, hi, 0, 1, 2, 3, 4, 5, 6, 7,
                                         8, 9, 10, 11, 12, 13, 14, 15);
}
__device__ __forceinline__ void wave_lds_sync() {
  __builtin_amdgcn_wave_barrier();
  asm volatile("s_wait_dscnt 0" ::: "memory");
  __builtin_amdgcn_wave_barrier();
}

// ---------------------------------------------------------------------------
// Prep 1: bulk f32 -> bf16 (8 elements / thread, b128 in, b128 out)
// ---------------------------------------------------------------------------
__global__ __launch_bounds__(256) void cvt_bf16_kernel(
    const float4* __restrict__ in, uint4* __restrict__ out, int n8) {
  const int i = blockIdx.x * blockDim.x + threadIdx.x;
  if (i >= n8) return;
  const float4 f0 = in[i * 2 + 0];
  const float4 f1 = in[i * 2 + 1];
  uint4 o;
  o.x = pk2bf(f0.x, f0.y);
  o.y = pk2bf(f0.z, f0.w);
  o.z = pk2bf(f1.x, f1.y);
  o.w = pk2bf(f1.z, f1.w);
  out[i] = o;
}

// ---------------------------------------------------------------------------
// Prep 2: W[1024][128] f32 -> Wt[128][1024] bf16 (x3 matrices via blockIdx.y)
// ---------------------------------------------------------------------------
__global__ __launch_bounds__(256) void wtrans_kernel(
    const float* __restrict__ Wq, const float* __restrict__ Wk,
    const float* __restrict__ Wv, unsigned short* __restrict__ wt) {
  const int sel = blockIdx.y;
  const float* W = (sel == 0) ? Wq : (sel == 1) ? Wk : Wv;
  unsigned short* out = wt + (size_t)sel * HEAD * EMBD;
  const int idx = blockIdx.x * blockDim.x + threadIdx.x;   // over HEAD*EMBD/8
  const int n  = idx / (EMBD / 8);
  const int k0 = (idx % (EMBD / 8)) * 8;
  v8u o;
#pragma unroll
  for (int j = 0; j < 8; ++j) o[j] = f2bf(W[(k0 + j) * HEAD + n]);
  *(v8u*)(out + (size_t)n * EMBD + k0) = o;
}

// ---------------------------------------------------------------------------
// Kernel 1: QKV projection, all-bf16 operands.  grid = (BT/16, 3), block 256.
// Each wave: one 16x16 tile, K=1024 in 32-steps; fragments are b128 pairs.
// Q pre-scaled by HEAD^-0.5; bias folded into accumulator init.
// V is stored transposed: vt[b][h][t] (for contiguous P*V B-fragments later).
// ---------------------------------------------------------------------------
__global__ __launch_bounds__(256) void qkv_kernel(
    const unsigned short* __restrict__ x_bf,
    const unsigned short* __restrict__ wt,
    const float* __restrict__ bq, const float* __restrict__ bk,
    const float* __restrict__ bv,
    unsigned short* __restrict__ qo,
    unsigned short* __restrict__ ko,
    unsigned short* __restrict__ vo) {
  const int sel = blockIdx.y;
  const float* bias = (sel == 0) ? bq : (sel == 1) ? bk : bv;
  const float scale = (sel == 0) ? 0.08838834764831845f : 1.0f;
  const unsigned short* Wt = wt + (size_t)sel * HEAD * EMBD;

  const int lane  = threadIdx.x & 31;
  const int wave  = threadIdx.x >> 5;
  const int col   = lane & 15;
  const int khalf = lane >> 4;
  const int m0    = blockIdx.x * 16;   // global row over B*T
  const int n0    = wave * 16;         // output column tile

  const float bval = bias[n0 + col];
  v8f c = { bval, bval, bval, bval, bval, bval, bval, bval };

  const unsigned short* xrow = x_bf + (size_t)(m0 + col) * EMBD;
  const unsigned short* wrow = Wt + (size_t)(n0 + col) * EMBD;

  for (int k0 = 0; k0 < EMBD; k0 += 32) {
    const v8u a0 = *(const v8u*)(xrow + k0 + khalf * 8);
    const v8u a1 = *(const v8u*)(xrow + k0 + 16 + khalf * 8);
    const v8u b0 = *(const v8u*)(wrow + k0 + khalf * 8);
    const v8u b1 = *(const v8u*)(wrow + k0 + 16 + khalf * 8);
    c = __builtin_amdgcn_wmma_f32_16x16x32_bf16(
        false, __builtin_bit_cast(v16bf, cat16(a0, a1)),
        false, __builtin_bit_cast(v16bf, cat16(b0, b1)),
        (short)0, c, false, false);
  }

  if (sel != 2) {
    unsigned short* out = (sel == 0) ? qo : ko;
#pragma unroll
    for (int r = 0; r < 8; ++r) {
      const int row = m0 + r + 8 * khalf;          // C-layout: VGPR r -> row
      out[(size_t)row * HEAD + n0 + col] = f2bf(c[r] * scale);
    }
  } else {
    const int bb = m0 / SEQT, t0 = m0 % SEQT;
    v8u o;
#pragma unroll
    for (int r = 0; r < 8; ++r) o[r] = f2bf(c[r]);
    // vt[b][h][t]; rows r -> consecutive t -> one b128 store
    *(v8u*)(vo + ((size_t)bb * HEAD + n0 + col) * SEQT + t0 + 8 * khalf) = o;
  }
}

// ---------------------------------------------------------------------------
// Kernel 2: causal flash attention.  grid = (T/64, B), block = 128 (4 waves).
// One wave per 16-query tile; keys in 32-wide blocks; S->P transpose + online
// softmax stats in wave-private LDS (DS ops are in-order per wave).
// ---------------------------------------------------------------------------
#define S_STRIDE  34
#define F_PER_W   (16 * S_STRIDE + 48)   // S[16][34] + m[16] + l[16] + alpha[16]
#define P_STRIDE  48                     // ushort stride (96B, 16B aligned)

__global__ __launch_bounds__(128) void attn_kernel(
    const unsigned short* __restrict__ q_bf,
    const unsigned short* __restrict__ k_bf,
    const unsigned short* __restrict__ vt_bf,
    float* __restrict__ out) {
  __shared__ __align__(16) float          ldsF[4 * F_PER_W];
  __shared__ __align__(16) unsigned short ldsP[4 * 16 * P_STRIDE];

  const int lane  = threadIdx.x & 31;
  const int wave  = threadIdx.x >> 5;
  const int col   = lane & 15;
  const int khalf = lane >> 4;
  const int b     = blockIdx.y;
  const int q0    = (blockIdx.x * 4 + wave) * 16;

  float* S    = ldsF + wave * F_PER_W;
  float* mrow = S + 16 * S_STRIDE;
  float* lrow = mrow + 16;
  float* arow = lrow + 16;
  unsigned short* Pw = ldsP + wave * 16 * P_STRIDE;

  if (lane < 16) { mrow[lane] = -1e30f; lrow[lane] = 0.0f; }
  wave_lds_sync();

  // Q A-fragments: 16(M) x 128(h) as four 16x32 chunks (pre-scaled)
  const unsigned short* qrow = q_bf + (size_t)(b * SEQT + q0 + col) * HEAD;
  v16bf aQ[4];
#pragma unroll
  for (int c4 = 0; c4 < 4; ++c4) {
    const v8u lo = *(const v8u*)(qrow + c4 * 32 + khalf * 8);
    const v8u hi = *(const v8u*)(qrow + c4 * 32 + 16 + khalf * 8);
    aQ[c4] = __builtin_bit_cast(v16bf, cat16(lo, hi));
  }

  const v8f vzero = { 0, 0, 0, 0, 0, 0, 0, 0 };
  v8f O[8];
#pragma unroll
  for (int nt = 0; nt < 8; ++nt) O[nt] = vzero;

  const int kend = q0 + 15;
  for (int kb = 0; kb <= kend; kb += 32) {
    if (kb + 32 <= kend)   // prefetch next key block (32 rows across 32 lanes)
      __builtin_prefetch(k_bf + (size_t)(b * SEQT + kb + 32 + lane) * HEAD, 0, 1);

    // S = Q K^T for this 16x32 key block (two 16x16 WMMA chains over h)
#pragma unroll
    for (int h2 = 0; h2 < 2; ++h2) {
      const int key = kb + h2 * 16 + col;   // B-frag: lane = key column of S
      const unsigned short* krow = k_bf + (size_t)(b * SEQT + key) * HEAD;
      v8f s = vzero;
#pragma unroll
      for (int c4 = 0; c4 < 4; ++c4) {
        const v8u lo = *(const v8u*)(krow + c4 * 32 + khalf * 8);
        const v8u hi = *(const v8u*)(krow + c4 * 32 + 16 + khalf * 8);
        s = __builtin_amdgcn_wmma_f32_16x16x32_bf16(
            false, aQ[c4], false, __builtin_bit_cast(v16bf, cat16(lo, hi)),
            (short)0, s, false, false);
      }
#pragma unroll
      for (int r = 0; r < 8; ++r)
        S[(r + 8 * khalf) * S_STRIDE + h2 * 16 + col] = s[r];
    }
    wave_lds_sync();

    // Online softmax: lanes 0..15 each own one query row; emit P as bf16
    if (lane < 16) {
      const int lr = lane;
      int ncols = q0 + lr - kb + 1;            // causal: keys <= query
      if (ncols > 32) ncols = 32;
      float alpha = 1.0f;
      unsigned int* prow = (unsigned int*)(Pw + lr * P_STRIDE);
      if (ncols <= 0) {
        for (int g = 0; g < 16; ++g) prow[g] = 0u;
      } else {
        const float m_old = mrow[lr];
        float rmax = -1e30f;
        for (int cc = 0; cc < ncols; ++cc) rmax = fmaxf(rmax, S[lr * S_STRIDE + cc]);
        const float m_new = fmaxf(m_old, rmax);
        alpha = __expf(m_old - m_new);
        float sum = 0.0f;
        for (int cc = 0; cc < 32; cc += 2) {
          const float p0 = (cc     < ncols) ? __expf(S[lr * S_STRIDE + cc]     - m_new) : 0.0f;
          const float p1 = (cc + 1 < ncols) ? __expf(S[lr * S_STRIDE + cc + 1] - m_new) : 0.0f;
          sum += p0 + p1;
          prow[cc >> 1] = pk2bf(p0, p1);
        }
        lrow[lr] = lrow[lr] * alpha + sum;
        mrow[lr] = m_new;
      }
      arow[lr] = alpha;
    }
    wave_lds_sync();

    // Rescale O by alpha of the rows this lane's accumulators cover
    float al[8];
#pragma unroll
    for (int r = 0; r < 8; ++r) al[r] = arow[r + 8 * khalf];
#pragma unroll
    for (int nt = 0; nt < 8; ++nt)
#pragma unroll
      for (int r = 0; r < 8; ++r) O[nt][r] *= al[r];

    // P A-fragment (16 queries x 32 keys): two ds_load_b128 per lane
    const v8u plo = *(const v8u*)(Pw + col * P_STRIDE + khalf * 8);
    const v8u phi = *(const v8u*)(Pw + col * P_STRIDE + 16 + khalf * 8);
    const v16bf P = __builtin_bit_cast(v16bf, cat16(plo, phi));

    // O += P * V  (8 n-tiles over h = 0..127); V^T rows -> contiguous b128
#pragma unroll
    for (int nt = 0; nt < 8; ++nt) {
      const int hcol = nt * 16 + col;
      const unsigned short* vrow =
          vt_bf + ((size_t)b * HEAD + hcol) * SEQT + kb;
      const v8u lo = *(const v8u*)(vrow + khalf * 8);
      const v8u hi = *(const v8u*)(vrow + 16 + khalf * 8);
      O[nt] = __builtin_amdgcn_wmma_f32_16x16x32_bf16(
          false, P, false, __builtin_bit_cast(v16bf, cat16(lo, hi)),
          (short)0, O[nt], false, false);
    }
  }

  wave_lds_sync();
  float linv[8];
#pragma unroll
  for (int r = 0; r < 8; ++r) linv[r] = 1.0f / lrow[r + 8 * khalf];

#pragma unroll
  for (int nt = 0; nt < 8; ++nt)
#pragma unroll
    for (int r = 0; r < 8; ++r) {
      const int row = q0 + r + 8 * khalf;
      out[(size_t)(b * SEQT + row) * HEAD + nt * 16 + col] = O[nt][r] * linv[r];
    }
}

// ---------------------------------------------------------------------------
extern "C" void kernel_launch(void* const* d_in, const int* in_sizes, int n_in,
                              void* d_out, int out_size, void* d_ws, size_t ws_size,
                              hipStream_t stream) {
  const float* x  = (const float*)d_in[0];
  const float* Wk = (const float*)d_in[1];
  const float* bk = (const float*)d_in[2];
  const float* Wq = (const float*)d_in[3];
  const float* bq = (const float*)d_in[4];
  const float* Wv = (const float*)d_in[5];
  const float* bv = (const float*)d_in[6];

  const size_t xe  = (size_t)NB * SEQT * EMBD;   // x elements
  const size_t we  = (size_t)HEAD * EMBD;        // one weight matrix
  const size_t pe  = (size_t)NB * SEQT * HEAD;   // one projection

  unsigned short* x_bf  = (unsigned short*)d_ws;
  unsigned short* wt    = x_bf + xe;             // 3 transposed weight matrices
  unsigned short* q_bf  = wt + 3 * we;
  unsigned short* k_bf  = q_bf + pe;
  unsigned short* vt_bf = k_bf + pe;             // transposed V

  const int n8 = (int)(xe / 8);
  cvt_bf16_kernel<<<dim3((n8 + 255) / 256), 256, 0, stream>>>(
      (const float4*)x, (uint4*)x_bf, n8);

  wtrans_kernel<<<dim3((HEAD * EMBD / 8) / 256, 3), 256, 0, stream>>>(
      Wq, Wk, Wv, wt);

  qkv_kernel<<<dim3((NB * SEQT) / 16, 3), 256, 0, stream>>>(
      x_bf, wt, bq, bk, bv, q_bf, k_bf, vt_bf);

  attn_kernel<<<dim3(SEQT / 64, NB), 128, 0, stream>>>(
      q_bf, k_bf, vt_bf, (float*)d_out);
}
